// hallucination_module_7799660609546
// MI455X (gfx1250) — matrix operand order repaired and checked by reading, not compiled, not run on hardware
//
#include <hip/hip_runtime.h>
#include <hip/hip_bf16.h>
#include <math.h>

typedef __attribute__((ext_vector_type(16))) __bf16 v16bf;
typedef __attribute__((ext_vector_type(2)))  __bf16 v2bf;
typedef __attribute__((ext_vector_type(8)))  float  v8f;

#define HW     65536      // 256*256
#define WIDTH  256
#define PAD_HW 257.0f     // clip max of padded coords (Hp-1 = 257)

// ---------------------------------------------------------------------------
// A-fragment element->K mapping for 16-bit A (16x32), contiguous: k = 16*half + e (matches B staging)
__device__ __forceinline__ int a_klocal(int e, int half) {
    return e + 16 * half;
}

// ---------------------------------------------------------------------------
// Kernel 1: f32 weights -> bf16 fragment-native layouts. K = n*64 + c.
//   wpm  : [18 kb][2 mt][32 lane][16 e]  rows 0..17=p_w, 18..26=m_w, 27..31=0
//   wmain: [18 kb][4 mt][32 lane][16 e]  rows = 64 out channels of w
__global__ __launch_bounds__(256) void prep_weights(
    const float* __restrict__ p_w, const float* __restrict__ m_w,
    const float* __restrict__ w,
    __bf16* __restrict__ wpm, __bf16* __restrict__ wmain)
{
    int idx = blockIdx.x * 256 + threadIdx.x;
    const int NPM = 18 * 2 * 32 * 16;   // 18432
    const int NMN = 18 * 4 * 32 * 16;   // 36864
    if (idx < NPM) {
        int e    = idx & 15;
        int lane = (idx >> 4) & 31;
        int mt   = (idx >> 9) & 1;
        int kb   = idx >> 10;
        int m    = mt * 16 + (lane & 15);
        int half = lane >> 4;
        int K    = kb * 32 + a_klocal(e, half);
        int n    = K >> 6, c = K & 63;
        float v  = 0.0f;
        if (m < 18)      v = p_w[(m * 64 + c) * 9 + n];
        else if (m < 27) v = m_w[((m - 18) * 64 + c) * 9 + n];
        wpm[idx] = (__bf16)v;
    } else if (idx < NPM + NMN) {
        int f    = idx - NPM;
        int e    = f & 15;
        int lane = (f >> 4) & 31;
        int mt   = (f >> 9) & 3;
        int kb   = f >> 11;
        int o    = mt * 16 + (lane & 15);
        int half = lane >> 4;
        int K    = kb * 32 + a_klocal(e, half);
        int n    = K >> 6, c = K & 63;
        wmain[f] = (__bf16)w[(o * 64 + c) * 9 + n];
    }
}

// ---------------------------------------------------------------------------
// Kernel 2: offset & mask convs as im2col GEMM. Block: 256 thr = 8 waves,
// 128 pixels x 32 out-rows. Wave: mt = w&1, column-tile pair ctp = w>>1.
// One A-fragment load feeds two WMMAs (B tiles ct0, ct1).
__global__ __launch_bounds__(256) void offset_mask_kernel(
    const float* __restrict__ x,
    const __bf16* __restrict__ wpm,
    const float* __restrict__ p_b, const float* __restrict__ m_b,
    float* __restrict__ out_offset, float* __restrict__ maskbuf)
{
    __shared__ __attribute__((aligned(32))) __bf16 lds_b[2][8 * 32 * 16]; // ping-pong

    const int t     = threadIdx.x;
    const int lane  = t & 31;
    const int wave  = t >> 5;
    const int mt    = wave & 1;
    const int ct0   = (wave >> 1) * 2, ct1 = ct0 + 1;
    const int pb    = blockIdx.x * 128;
    const int row   = pb >> 8;       // 128 | 256 -> single image row per block
    const int pbcol = pb & 255;

    v8f acc0 = {}, acc1 = {};

    for (int n = 0; n < 9; ++n) {
        const int i  = n / 3;
        const int j  = n - 3 * i;
        const int sr = row + 2 * i - 2;
        const int sc0base = pbcol + 2 * j - 2;
        const bool row_ok = (unsigned)sr < 256u;
        const float* __restrict__ xrow = x + sr * WIDTH;

        #pragma unroll
        for (int h2 = 0; h2 < 2; ++h2) {
            const int kb  = n * 2 + h2;
            const int cb  = h2 * 32;
            const int buf = h2;            // kb & 1 == h2

            // ---- stage: 2px x 2ch per task; float2 loads, packed b32 LDS stores
            #pragma unroll
            for (int qq = 0; qq < 4; ++qq) {
                int idxq = qq * 256 + t;        // 0..1023
                int kp   = idxq >> 6;           // 0..15 channel pair
                int pp   = idxq & 63;           // 0..63 pixel pair
                int Kl   = kp * 2;
                int pl   = pp * 2;
                int c    = cb + Kl;
                int sc   = sc0base + pl;        // even -> float2 aligned
                bool ok  = row_ok && ((unsigned)sc < 256u);
                int scs  = ok ? sc : 0;
                float2 va = *(const float2*)(xrow + c * HW + scs);
                float2 vb = *(const float2*)(xrow + (c + 1) * HW + scs);
                if (!ok) { va.x = 0.f; va.y = 0.f; vb.x = 0.f; vb.y = 0.f; }
                int half = Kl >> 4, e = Kl & 15;        // e even
                int bt = pl >> 4, col = pl & 15;        // col even
                int base = ((bt * 32 + half * 16 + col) << 4) + e;
                v2bf w0 = { (__bf16)va.x, (__bf16)vb.x };   // pixel pl
                v2bf w1 = { (__bf16)va.y, (__bf16)vb.y };   // pixel pl+1
                *(v2bf*)(&lds_b[buf][base])      = w0;
                *(v2bf*)(&lds_b[buf][base + 16]) = w1;
            }
            __syncthreads();

            v16bf a  = *(const v16bf*)(wpm + (((kb * 2 + mt) * 32 + lane) << 4));
            v16bf b0 = *(const v16bf*)(&lds_b[buf][(ct0 * 32 + lane) << 4]);
            v16bf b1 = *(const v16bf*)(&lds_b[buf][(ct1 * 32 + lane) << 4]);
            acc0 = __builtin_amdgcn_wmma_f32_16x16x32_bf16(
                       false, a, false, b0, (short)0, acc0, false, false);
            acc1 = __builtin_amdgcn_wmma_f32_16x16x32_bf16(
                       false, a, false, b1, (short)0, acc1, false, false);
            // ping-pong: no trailing barrier
        }
    }

    // ---- epilogue: bias (+ sigmoid for mask rows), both column tiles
    const int col  = lane & 15;
    const int half = lane >> 4;
    #pragma unroll
    for (int r = 0; r < 8; ++r) {
        int ch = mt * 16 + r + 8 * half;
        float v0 = acc0[r], v1 = acc1[r];
        int p0 = pb + ct0 * 16 + col;
        int p1 = pb + ct1 * 16 + col;
        if (ch < 18) {
            float bias = p_b[ch];
            out_offset[ch * HW + p0] = v0 + bias;
            out_offset[ch * HW + p1] = v1 + bias;
        } else if (ch < 27) {
            float bias = m_b[ch - 18];
            maskbuf[(ch - 18) * HW + p0] = 1.0f / (1.0f + __expf(-(v0 + bias)));
            maskbuf[(ch - 18) * HW + p1] = 1.0f / (1.0f + __expf(-(v1 + bias)));
        }
    }
}

// ---------------------------------------------------------------------------
// Kernel 3: modulated deformable conv. Block: 256 thr = 8 waves,
// 64 pixels x 64 out channels. Wave: mt = w&3, column-tile pair ctp = w>>2.
// Gathers are branchless: OOB taps have weight 0 and offset 0.
__global__ __launch_bounds__(256) void deform_kernel(
    const float* __restrict__ x,
    const __bf16* __restrict__ wmain,
    const float* __restrict__ offs,      // [18][HW]
    const float* __restrict__ maskbuf,   // [9][HW]
    float* __restrict__ out)             // [64][HW]
{
    __shared__ float4 g_lds[576];        // [n][pl] mask-folded weights (0 if OOB)
    __shared__ int4   o_lds[576];        // [n][pl] gather offsets (0 if OOB)
    __shared__ __attribute__((aligned(32))) __bf16 lds_b[2][4 * 32 * 16];

    const int t    = threadIdx.x;
    const int lane = t & 31;
    const int wave = t >> 5;
    const int mt   = wave & 3;
    const int ct0  = (wave >> 2) * 2, ct1 = ct0 + 1;
    const int pb   = blockIdx.x * 64;

    // ---- precompute per-(tap, pixel) sampling data (576 entries)
    for (int s = t; s < 576; s += 256) {
        int pl = s & 63, n = s >> 6;     // s = n*64 + pl
        int p  = pb + pl;
        int prow = p >> 8, pcol = p & 255;
        int i = n / 3, j = n - 3 * i;
        float px = (float)(prow + i) + offs[n * HW + p];
        float py = (float)(pcol + j) + offs[(9 + n) * HW + p];
        float fx = floorf(px), fy = floorf(py);
        float qltx = fminf(fmaxf(fx,       0.f), PAD_HW);
        float qlty = fminf(fmaxf(fy,       0.f), PAD_HW);
        float qrbx = fminf(fmaxf(fx + 1.f, 0.f), PAD_HW);
        float qrby = fminf(fmaxf(fy + 1.f, 0.f), PAD_HW);
        float pxc  = fminf(fmaxf(px,       0.f), PAD_HW);
        float pyc  = fminf(fmaxf(py,       0.f), PAD_HW);
        float gxl = 1.f + (qltx - pxc), gxr = 1.f - (qrbx - pxc);
        float gyl = 1.f + (qlty - pyc), gyr = 1.f - (qrby - pyc);
        float m = maskbuf[n * HW + p];
        int ix0 = (int)qltx, iy0 = (int)qlty;
        int ix1 = (int)qrbx, iy1 = (int)qrby;
        bool vx0 = (ix0 >= 1 && ix0 <= 256), vy0 = (iy0 >= 1 && iy0 <= 256);
        bool vx1 = (ix1 >= 1 && ix1 <= 256), vy1 = (iy1 >= 1 && iy1 <= 256);
        float4 g;
        int4 o;
        g.x = (vx0 && vy0) ? gxl * gyl * m : 0.f;
        g.y = (vx1 && vy1) ? gxr * gyr * m : 0.f;
        g.z = (vx0 && vy1) ? gxl * gyr * m : 0.f;
        g.w = (vx1 && vy0) ? gxr * gyl * m : 0.f;
        o.x = (vx0 && vy0) ? ((ix0 - 1) * WIDTH + (iy0 - 1)) : 0;
        o.y = (vx1 && vy1) ? ((ix1 - 1) * WIDTH + (iy1 - 1)) : 0;
        o.z = (vx0 && vy1) ? ((ix0 - 1) * WIDTH + (iy1 - 1)) : 0;
        o.w = (vx1 && vy0) ? ((ix1 - 1) * WIDTH + (iy0 - 1)) : 0;
        g_lds[s] = g;
        o_lds[s] = o;
    }
    __syncthreads();

    v8f acc0 = {}, acc1 = {};

    const int pl   = t & 63;             // pixel this thread samples for
    const int krow = t >> 6;             // 0..3 -> 8 channels per thread
    const int bt   = pl >> 4, bcol = pl & 15;
    const int base0 = ((bt * 32 + ((bcol) ? 0 : 0)) << 0); (void)base0;

    for (int n = 0; n < 9; ++n) {
        float4 g = g_lds[(n << 6) + pl];     // one metadata read per tap
        int4   o = o_lds[(n << 6) + pl];

        #pragma unroll
        for (int h2 = 0; h2 < 2; ++h2) {
            const int kb  = n * 2 + h2;
            const int cb  = h2 * 32;
            const int buf = h2;

            #pragma unroll
            for (int qp = 0; qp < 4; ++qp) {
                int Kl = krow * 8 + qp * 2;      // even
                int c  = cb + Kl;
                const float* xp0 = x + c * HW;
                const float* xp1 = xp0 + HW;
                float v0 = g.x * xp0[o.x] + g.y * xp0[o.y]
                         + g.z * xp0[o.z] + g.w * xp0[o.w];
                float v1 = g.x * xp1[o.x] + g.y * xp1[o.y]
                         + g.z * xp1[o.z] + g.w * xp1[o.w];
                int half = Kl >> 4, e = Kl & 15; // e even
                int idx0 = ((bt * 32 + half * 16 + bcol) << 4) + e;
                v2bf wv = { (__bf16)v0, (__bf16)v1 };   // channels c, c+1
                *(v2bf*)(&lds_b[buf][idx0]) = wv;
            }
            __syncthreads();

            v16bf a  = *(const v16bf*)(wmain + (((kb * 4 + mt) * 32 + lane) << 4));
            v16bf b0 = *(const v16bf*)(&lds_b[buf][(ct0 * 32 + lane) << 4]);
            v16bf b1 = *(const v16bf*)(&lds_b[buf][(ct1 * 32 + lane) << 4]);
            acc0 = __builtin_amdgcn_wmma_f32_16x16x32_bf16(
                       false, a, false, b0, (short)0, acc0, false, false);
            acc1 = __builtin_amdgcn_wmma_f32_16x16x32_bf16(
                       false, a, false, b1, (short)0, acc1, false, false);
            // ping-pong: no trailing barrier
        }
    }

    // ---- epilogue
    const int col  = lane & 15;
    const int half = lane >> 4;
    #pragma unroll
    for (int r = 0; r < 8; ++r) {
        int ch = mt * 16 + r + 8 * half;
        out[ch * HW + pb + ct0 * 16 + col] = acc0[r];
        out[ch * HW + pb + ct1 * 16 + col] = acc1[r];
    }
}

// ---------------------------------------------------------------------------
extern "C" void kernel_launch(void* const* d_in, const int* in_sizes, int n_in,
                              void* d_out, int out_size, void* d_ws, size_t ws_size,
                              hipStream_t stream) {
    const float* x   = (const float*)d_in[0];
    const float* p_w = (const float*)d_in[1];
    const float* p_b = (const float*)d_in[2];
    const float* m_w = (const float*)d_in[3];
    const float* m_b = (const float*)d_in[4];
    const float* w   = (const float*)d_in[5];

    float* out        = (float*)d_out;            // [64][65536]
    float* out_offset = out + 64 * HW;            // [18][65536] (second return value)

    __bf16* wpm   = (__bf16*)d_ws;                // 18*2*32*16 bf16
    __bf16* wmain = wpm + 18 * 2 * 32 * 16;       // 18*4*32*16 bf16
    float*  maskbuf = (float*)(wmain + 18 * 4 * 32 * 16);  // [9][65536] f32

    prep_weights<<<216, 256, 0, stream>>>(p_w, m_w, w, wpm, wmain);

    // 65536 px / 128 px per block
    offset_mask_kernel<<<512, 256, 0, stream>>>(x, wpm, p_b, m_b,
                                                out_offset, maskbuf);

    // 65536 px / 64 px per block
    deform_kernel<<<1024, 256, 0, stream>>>(x, wmain, out_offset, maskbuf, out);
}
